// MultiheadAttentionBlock_41996190220540
// MI455X (gfx1250) — compile-verified, matrix operand order checked by reading, not run on hardware
//
#include <hip/hip_runtime.h>
#include <hip/hip_bf16.h>
#include <math.h>

typedef __attribute__((ext_vector_type(16))) __bf16 bf16x16;
typedef __attribute__((ext_vector_type(8)))  __bf16 bf16x8;
typedef __attribute__((ext_vector_type(8)))  float  floatx8;

#define DIMV 512
#define NB   8
#define NSQ  1024
#define NSK  1024
#define NH   8
#define HD   64
#define NEG_INF (-1.0e38f)

// GEMM LDS tile row stride (32 K + 16 pad), attention tile row stride (64 + 16)
#define GSTRIDE 48
#define ASTRIDE 80

__device__ __forceinline__ __bf16 f2bf(float f) { return (__bf16)f; }

// Fragment element e -> K = koff + e (e<8), 16 + koff + (e-8) (e>=8), koff = 8*(lane>>4).
// So a fragment is two contiguous 8-element (16B) runs at rowbase+koff and rowbase+16+koff.
__device__ __forceinline__ bf16x16 ld_frag(const __bf16* rowbase, int koff) {
  union { struct { bf16x8 lo, hi; } p; bf16x16 v; } u;
  u.p.lo = *(const bf16x8*)(rowbase + koff);
  u.p.hi = *(const bf16x8*)(rowbase + 16 + koff);
  return u.v;
}

// -------------------------------------------------------------------------
// Generic bf16-WMMA GEMM:  C[M,N] = A[M,K] (fp32) * W[K,N] (fp32) + bias
// epilogue:
//   if resid: v = resid[idx] + relu(acc + bias)  else v = acc + bias
//   if outF:  outF[idx] = v
//   if outB:  outB[idx] = (bf16)(v * bscale)
// Tiles: 128x128x32, 256 threads (8 waves, 4(M) x 2(N)), wave = 32x64.
// A staged row-major, W staged TRANSPOSED so both operands read as 2x b128.
// -------------------------------------------------------------------------
__global__ void __launch_bounds__(256)
gemm_wmma_bf16(const float* __restrict__ A, const float* __restrict__ W,
               const float* __restrict__ bias,
               float* outF, __bf16* outB, const float* resid,
               int M, int N, int K, float bscale)
{
  __shared__ __bf16 As [128][GSTRIDE];   // [m][k]
  __shared__ __bf16 Bst[128][GSTRIDE];   // [n][k]  (W transposed)

  const int tid  = threadIdx.x;
  const int lane = tid & 31;
  const int wave = tid >> 5;
  const int wm   = wave & 3;           // 0..3  (M dir)
  const int wn   = wave >> 2;          // 0..1  (N dir)
  const int ln   = lane & 15;
  const int koff = (lane >> 4) << 3;   // 8*half
  const int bm0  = blockIdx.y * 128;
  const int bn0  = blockIdx.x * 128;

  floatx8 acc[2][4];
  const floatx8 zero8 = {0.f,0.f,0.f,0.f,0.f,0.f,0.f,0.f};
#pragma unroll
  for (int mt = 0; mt < 2; ++mt)
#pragma unroll
    for (int nt = 0; nt < 4; ++nt) acc[mt][nt] = zero8;

  for (int k0 = 0; k0 < K; k0 += 32) {
    __syncthreads();
    // ---- stage A tile (128x32): 8-elem chunks, vector load + vector LDS store
#pragma unroll
    for (int c = 0; c < 2; ++c) {
      int idx = (tid + c * 256) << 3;         // element index in 128x32
      int r = idx >> 5, col = idx & 31;
      const float* src = A + (size_t)(bm0 + r) * K + (k0 + col);
      float4 f0 = *(const float4*)(src);
      float4 f1 = *(const float4*)(src + 4);
      union { __bf16 h[8]; bf16x8 v; } pk;
      pk.h[0]=f2bf(f0.x); pk.h[1]=f2bf(f0.y); pk.h[2]=f2bf(f0.z); pk.h[3]=f2bf(f0.w);
      pk.h[4]=f2bf(f1.x); pk.h[5]=f2bf(f1.y); pk.h[6]=f2bf(f1.z); pk.h[7]=f2bf(f1.w);
      *(bf16x8*)&As[r][col] = pk.v;
      if (k0 + 32 < K) __builtin_prefetch(src + 32, 0, 1);
    }
    // ---- stage W tile transposed (32x128 -> Bst[n][k]): vector global load,
    //      8 scattered b16 stores
#pragma unroll
    for (int c = 0; c < 2; ++c) {
      int idx = (tid + c * 256) << 3;         // element index in 32x128
      int r = idx >> 7, col = idx & 127;
      const float* src = W + (size_t)(k0 + r) * N + (bn0 + col);
      float4 f0 = *(const float4*)(src);
      float4 f1 = *(const float4*)(src + 4);
      Bst[col + 0][r] = f2bf(f0.x);
      Bst[col + 1][r] = f2bf(f0.y);
      Bst[col + 2][r] = f2bf(f0.z);
      Bst[col + 3][r] = f2bf(f0.w);
      Bst[col + 4][r] = f2bf(f1.x);
      Bst[col + 5][r] = f2bf(f1.y);
      Bst[col + 6][r] = f2bf(f1.z);
      Bst[col + 7][r] = f2bf(f1.w);
    }
    __syncthreads();

    bf16x16 af[2], bfr[4];
#pragma unroll
    for (int mt = 0; mt < 2; ++mt)
      af[mt] = ld_frag(&As[wm * 32 + mt * 16 + ln][0], koff);
#pragma unroll
    for (int nt = 0; nt < 4; ++nt)
      bfr[nt] = ld_frag(&Bst[wn * 64 + nt * 16 + ln][0], koff);
#pragma unroll
    for (int mt = 0; mt < 2; ++mt)
#pragma unroll
      for (int nt = 0; nt < 4; ++nt)
        acc[mt][nt] = __builtin_amdgcn_wmma_f32_16x16x32_bf16(
            false, af[mt], false, bfr[nt], (short)0, acc[mt][nt], false, false);
  }

  // Epilogue
#pragma unroll
  for (int mt = 0; mt < 2; ++mt) {
#pragma unroll
    for (int nt = 0; nt < 4; ++nt) {
      int col = bn0 + wn * 64 + nt * 16 + ln;
      float bv = bias ? bias[col] : 0.f;
#pragma unroll
      for (int r = 0; r < 8; ++r) {
        int row = bm0 + wm * 32 + mt * 16 + r + koff;
        float v = acc[mt][nt][r] + bv;
        size_t idx = (size_t)row * N + col;
        if (resid) v = resid[idx] + fmaxf(v, 0.f);
        if (outF) outF[idx] = v;
        if (outB) outB[idx] = f2bf(v * bscale);
      }
    }
  }
}

// -------------------------------------------------------------------------
// Flash-style attention: one block = (b,h) x 64 query rows. 4 waves, each
// wave owns 16 q-rows. S = Q*K^T via WMMA (logit scale pre-folded into Qbf),
// presence masking, online softmax, O += P*V via WMMA (P via LDS).
// K staged NATURALLY [key][d] (that IS fragment layout for S's B operand);
// V staged transposed [d][key] for PV's B operand.
// Output: o1 = q_f32 + softmax(S) * V
// -------------------------------------------------------------------------
__global__ void __launch_bounds__(128)
attn_wmma(const __bf16* __restrict__ qb, const __bf16* __restrict__ kb,
          const __bf16* __restrict__ vb, const float* __restrict__ qf,
          const float* __restrict__ pres_q, const float* __restrict__ pres_k,
          float* __restrict__ o1)
{
  __shared__ __bf16 Qs[64][ASTRIDE];      // [qrow][d]
  __shared__ __bf16 Ks[64][ASTRIDE];      // [key][d]
  __shared__ __bf16 VT[64][ASTRIDE];      // [d][key]
  __shared__ __bf16 Ps[4][16][64];        // per-wave P, row stride 128B

  const int bh   = blockIdx.y;
  const int b    = bh >> 3;
  const int h    = bh & 7;
  const int q0   = blockIdx.x * 64;
  const int tid  = threadIdx.x;
  const int wave = tid >> 5;
  const int lane = tid & 31;
  const int ln   = lane & 15;
  const int koff = (lane >> 4) << 3;   // 8*half

  // Stage Q tile (already scaled by 1/sqrt(512) at projection time)
#pragma unroll
  for (int c = 0; c < 4; ++c) {
    int idx = (tid + c * 128) << 3;     // element in 64x64
    int r = idx >> 6, col = idx & 63;
    *(bf16x8*)&Qs[r][col] =
        *(const bf16x8*)&qb[((size_t)b * NSQ + q0 + r) * DIMV + h * HD + col];
  }
  __syncthreads();

  bf16x16 aq[2];
#pragma unroll
  for (int ks = 0; ks < 2; ++ks)
    aq[ks] = ld_frag(&Qs[wave * 16 + ln][ks * 32], koff);

  float pqv[8];
#pragma unroll
  for (int r = 0; r < 8; ++r)
    pqv[r] = pres_q[(size_t)b * NSQ + q0 + wave * 16 + r + koff];

  const floatx8 zero8 = {0.f,0.f,0.f,0.f,0.f,0.f,0.f,0.f};
  float m[8], l[8];
  floatx8 o[4];
#pragma unroll
  for (int r = 0; r < 8; ++r) { m[r] = NEG_INF; l[r] = 0.f; }
#pragma unroll
  for (int dt = 0; dt < 4; ++dt) o[dt] = zero8;

  for (int kk = 0; kk < NSK; kk += 64) {
    // Stage K (natural) and V (transposed) tiles
#pragma unroll
    for (int c = 0; c < 4; ++c) {
      int idx = (tid + c * 128) << 3;
      int key = idx >> 6, dd = idx & 63;
      size_t src = ((size_t)b * NSK + kk + key) * DIMV + h * HD + dd;
      *(bf16x8*)&Ks[key][dd] = *(const bf16x8*)&kb[src];
      union { __bf16 h[8]; bf16x8 v; } pv;
      pv.v = *(const bf16x8*)&vb[src];
      VT[dd + 0][key] = pv.h[0];
      VT[dd + 1][key] = pv.h[1];
      VT[dd + 2][key] = pv.h[2];
      VT[dd + 3][key] = pv.h[3];
      VT[dd + 4][key] = pv.h[4];
      VT[dd + 5][key] = pv.h[5];
      VT[dd + 6][key] = pv.h[6];
      VT[dd + 7][key] = pv.h[7];
    }
    __syncthreads();

    // S = Q * K^T   (B operand column = key -> read Ks[key][*] rows)
    floatx8 s[4];
    float pkv[4];
#pragma unroll
    for (int nt = 0; nt < 4; ++nt) {
      s[nt] = zero8;
#pragma unroll
      for (int ks = 0; ks < 2; ++ks) {
        bf16x16 bk = ld_frag(&Ks[nt * 16 + ln][ks * 32], koff);
        s[nt] = __builtin_amdgcn_wmma_f32_16x16x32_bf16(
            false, aq[ks], false, bk, (short)0, s[nt], false, false);
      }
      pkv[nt] = pres_k[(size_t)b * NSK + kk + nt * 16 + ln];
    }

    // mask + online softmax (rows live in 16-lane halves -> xor 1,2,4,8)
#pragma unroll
    for (int r = 0; r < 8; ++r) {
      float rm = NEG_INF;
#pragma unroll
      for (int nt = 0; nt < 4; ++nt) {
        float sv = ((pkv[nt] > 0.f) && (pqv[r] > 0.f)) ? s[nt][r] : NEG_INF;
        s[nt][r] = sv;
        rm = fmaxf(rm, sv);
      }
      rm = fmaxf(rm, __shfl_xor(rm, 1));
      rm = fmaxf(rm, __shfl_xor(rm, 2));
      rm = fmaxf(rm, __shfl_xor(rm, 4));
      rm = fmaxf(rm, __shfl_xor(rm, 8));
      float mn    = fmaxf(m[r], rm);
      float alpha = __expf(m[r] - mn);
      float rsum  = 0.f;
#pragma unroll
      for (int nt = 0; nt < 4; ++nt) {
        float p = __expf(s[nt][r] - mn);
        s[nt][r] = p;
        rsum += p;
      }
      rsum += __shfl_xor(rsum, 1);
      rsum += __shfl_xor(rsum, 2);
      rsum += __shfl_xor(rsum, 4);
      rsum += __shfl_xor(rsum, 8);
      l[r] = l[r] * alpha + rsum;
      m[r] = mn;
#pragma unroll
      for (int dt = 0; dt < 4; ++dt) o[dt][r] *= alpha;
    }

    // P -> LDS (C layout -> A layout round trip; scatter is inherent)
#pragma unroll
    for (int nt = 0; nt < 4; ++nt)
#pragma unroll
      for (int r = 0; r < 8; ++r)
        Ps[wave][r + koff][nt * 16 + ln] = f2bf(s[nt][r]);
    __syncthreads();

    // O += P * V   (B operand column = d -> read VT[d][*] rows)
#pragma unroll
    for (int dt = 0; dt < 4; ++dt) {
#pragma unroll
      for (int ks = 0; ks < 2; ++ks) {
        bf16x16 ap  = ld_frag(&Ps[wave][ln][ks * 32], koff);
        bf16x16 bv_ = ld_frag(&VT[dt * 16 + ln][ks * 32], koff);
        o[dt] = __builtin_amdgcn_wmma_f32_16x16x32_bf16(
            false, ap, false, bv_, (short)0, o[dt], false, false);
      }
    }
    __syncthreads();
  }

  // o1 = q (residual) + O / l
#pragma unroll
  for (int dt = 0; dt < 4; ++dt) {
#pragma unroll
    for (int r = 0; r < 8; ++r) {
      int row = q0 + wave * 16 + r + koff;
      int col = h * HD + dt * 16 + ln;
      size_t idx = ((size_t)b * NSQ + row) * DIMV + col;
      o1[idx] = o[dt][r] / l[r] + qf[idx];
    }
  }
}

// -------------------------------------------------------------------------
// LayerNorm over last dim (512). One block per row, 256 threads x 2 elems.
// -------------------------------------------------------------------------
__global__ void __launch_bounds__(256)
layernorm512(const float* __restrict__ x, const float* __restrict__ g,
             const float* __restrict__ bb, float* __restrict__ y)
{
  __shared__ float red[256];
  const size_t row = blockIdx.x;
  const int tid = threadIdx.x;
  const float* xr = x + row * DIMV;
  float v0 = xr[tid];
  float v1 = xr[tid + 256];

  red[tid] = v0 + v1;
  __syncthreads();
#pragma unroll
  for (int s = 128; s > 0; s >>= 1) {
    if (tid < s) red[tid] += red[tid + s];
    __syncthreads();
  }
  float mu = red[0] * (1.f / 512.f);
  __syncthreads();

  float d0 = v0 - mu, d1 = v1 - mu;
  red[tid] = d0 * d0 + d1 * d1;
  __syncthreads();
#pragma unroll
  for (int s = 128; s > 0; s >>= 1) {
    if (tid < s) red[tid] += red[tid + s];
    __syncthreads();
  }
  float inv = rsqrtf(red[0] * (1.f / 512.f) + 1e-5f);

  y[row * DIMV + tid]       = d0 * inv * g[tid]       + bb[tid];
  y[row * DIMV + tid + 256] = d1 * inv * g[tid + 256] + bb[tid + 256];
}

// -------------------------------------------------------------------------
extern "C" void kernel_launch(void* const* d_in, const int* in_sizes, int n_in,
                              void* d_out, int out_size, void* d_ws, size_t ws_size,
                              hipStream_t stream)
{
  const float* queries = (const float*)d_in[0];
  const float* keys    = (const float*)d_in[1];
  const float* pres_q  = (const float*)d_in[2];
  const float* pres_k  = (const float*)d_in[3];
  // d_in[4] = num_heads (compile-time NH)
  const float* Wq = (const float*)d_in[5];
  const float* bq = (const float*)d_in[6];
  const float* Wk = (const float*)d_in[7];
  const float* bk = (const float*)d_in[8];
  const float* Wv = (const float*)d_in[9];
  const float* bv = (const float*)d_in[10];
  const float* Wo = (const float*)d_in[11];
  const float* bo = (const float*)d_in[12];
  const float* g0 = (const float*)d_in[13];
  const float* b0 = (const float*)d_in[14];
  const float* g1 = (const float*)d_in[15];
  const float* b1 = (const float*)d_in[16];
  float* out = (float*)d_out;

  const size_t MROWS = (size_t)NB * NSQ;   // 8192
  char* ws = (char*)d_ws;
  __bf16* q_bf = (__bf16*)ws;  ws += MROWS * DIMV * sizeof(__bf16);
  __bf16* k_bf = (__bf16*)ws;  ws += MROWS * DIMV * sizeof(__bf16);
  __bf16* v_bf = (__bf16*)ws;  ws += MROWS * DIMV * sizeof(__bf16);
  float*  q_f  = (float*)ws;   ws += MROWS * DIMV * sizeof(float);   // q resid, later ln0
  float*  o1   = (float*)ws;   ws += MROWS * DIMV * sizeof(float);   // attn out, later ffn out

  const float logit_scale = 1.0f / sqrtf((float)DIMV);

  dim3 gg(DIMV / 128, MROWS / 128);   // (4, 64)
  // Q/K/V projections (Q bf16 copy pre-scaled by 1/sqrt(dim_v))
  gemm_wmma_bf16<<<gg, 256, 0, stream>>>(queries, Wq, bq, q_f, q_bf, nullptr,
                                         (int)MROWS, DIMV, DIMV, logit_scale);
  gemm_wmma_bf16<<<gg, 256, 0, stream>>>(keys, Wk, bk, nullptr, k_bf, nullptr,
                                         (int)MROWS, DIMV, DIMV, 1.0f);
  gemm_wmma_bf16<<<gg, 256, 0, stream>>>(keys, Wv, bv, nullptr, v_bf, nullptr,
                                         (int)MROWS, DIMV, DIMV, 1.0f);
  // Attention + residual
  attn_wmma<<<dim3(NSQ / 64, NB * NH), 128, 0, stream>>>(
      q_bf, k_bf, v_bf, q_f, pres_q, pres_k, o1);
  // LN0 (into q_f, reused)
  layernorm512<<<(unsigned)MROWS, 256, 0, stream>>>(o1, g0, b0, q_f);
  // o = ln0 + relu(ln0 @ Wo + bo)  (into o1, reused)
  gemm_wmma_bf16<<<gg, 256, 0, stream>>>(q_f, Wo, bo, o1, nullptr, q_f,
                                         (int)MROWS, DIMV, DIMV, 1.0f);
  // LN1 -> final output
  layernorm512<<<(unsigned)MROWS, 256, 0, stream>>>(o1, g1, b1, out);
}
